// GaussianSplattingData_721554505936
// MI455X (gfx1250) — compile-verified, matrix operand order checked by reading, not run on hardware
//
#include <hip/hip_runtime.h>
#include <hip/hip_bf16.h>
#include <math.h>

#define HH 192
#define WW 192
#define NG 512
#define EPSF 1e-6f
#define PSTRIDE 12   // u v Ac Bc Cc op bt r g b pad pad

typedef __attribute__((ext_vector_type(2)))  float    v2f;
typedef __attribute__((ext_vector_type(8)))  float    v8f;
typedef __attribute__((ext_vector_type(16))) _Float16 v16h;

// ---------------------------------------------------------------------------
// Kernel 1: per-gaussian projection / conic setup
// ---------------------------------------------------------------------------
__global__ void gs_preprocess(const float* __restrict__ means,
                              const float* __restrict__ quats,
                              const float* __restrict__ scales,
                              const float* __restrict__ opac,
                              const float* __restrict__ colors,
                              const float* __restrict__ betas,
                              const float* __restrict__ viewmats,
                              const float* __restrict__ Ks,
                              float* __restrict__ keys,
                              float* __restrict__ params, int N)
{
    int n = blockIdx.x * blockDim.x + threadIdx.x;
    if (n >= N) return;

    float R[3][3], t[3];
    #pragma unroll
    for (int i = 0; i < 3; ++i) {
        #pragma unroll
        for (int j = 0; j < 3; ++j) R[i][j] = viewmats[i * 4 + j];
        t[i] = viewmats[i * 4 + 3];
    }
    float fx = Ks[0], cx = Ks[2], fy = Ks[4], cy = Ks[5];

    float m0 = means[n * 3 + 0], m1 = means[n * 3 + 1], m2 = means[n * 3 + 2];
    float px = R[0][0] * m0 + R[0][1] * m1 + R[0][2] * m2 + t[0];
    float py = R[1][0] * m0 + R[1][1] * m1 + R[1][2] * m2 + t[1];
    float pz = R[2][0] * m0 + R[2][1] * m1 + R[2][2] * m2 + t[2];

    // quaternion -> rotation
    float qw = quats[n * 4 + 0], qx = quats[n * 4 + 1];
    float qy = quats[n * 4 + 2], qz = quats[n * 4 + 3];
    float qn = rsqrtf(qw * qw + qx * qx + qy * qy + qz * qz);
    qw *= qn; qx *= qn; qy *= qn; qz *= qn;
    float Rq[3][3];
    Rq[0][0] = 1.f - 2.f * (qy * qy + qz * qz);
    Rq[0][1] = 2.f * (qx * qy - qw * qz);
    Rq[0][2] = 2.f * (qx * qz + qw * qy);
    Rq[1][0] = 2.f * (qx * qy + qw * qz);
    Rq[1][1] = 1.f - 2.f * (qx * qx + qz * qz);
    Rq[1][2] = 2.f * (qy * qz - qw * qx);
    Rq[2][0] = 2.f * (qx * qz - qw * qy);
    Rq[2][1] = 2.f * (qy * qz + qw * qx);
    Rq[2][2] = 1.f - 2.f * (qx * qx + qy * qy);

    float e0 = __expf(scales[n * 3 + 0]);
    float e1 = __expf(scales[n * 3 + 1]);
    float e2 = __expf(scales[n * 3 + 2]);
    float M[3][3];
    #pragma unroll
    for (int i = 0; i < 3; ++i) { M[i][0] = Rq[i][0] * e0; M[i][1] = Rq[i][1] * e1; M[i][2] = Rq[i][2] * e2; }

    float C3[3][3];            // M M^T
    #pragma unroll
    for (int i = 0; i < 3; ++i)
        #pragma unroll
        for (int k = 0; k < 3; ++k)
            C3[i][k] = M[i][0] * M[k][0] + M[i][1] * M[k][1] + M[i][2] * M[k][2];

    float TT[3][3], covc[3][3];  // covc = R C3 R^T
    #pragma unroll
    for (int i = 0; i < 3; ++i)
        #pragma unroll
        for (int k = 0; k < 3; ++k)
            TT[i][k] = R[i][0] * C3[0][k] + R[i][1] * C3[1][k] + R[i][2] * C3[2][k];
    #pragma unroll
    for (int i = 0; i < 3; ++i)
        #pragma unroll
        for (int l = 0; l < 3; ++l)
            covc[i][l] = TT[i][0] * R[l][0] + TT[i][1] * R[l][1] + TT[i][2] * R[l][2];

    float tz = fmaxf(pz, 0.01f);
    float z1 = 1.f / tz;
    float j00 = fx * z1, j02 = -fx * px * z1 * z1;
    float j11 = fy * z1, j12 = -fy * py * z1 * z1;

    float JC00 = j00 * covc[0][0] + j02 * covc[2][0];
    float JC01 = j00 * covc[0][1] + j02 * covc[2][1];
    float JC02 = j00 * covc[0][2] + j02 * covc[2][2];
    float JC10 = j11 * covc[1][0] + j12 * covc[2][0];
    float JC11 = j11 * covc[1][1] + j12 * covc[2][1];
    float JC12 = j11 * covc[1][2] + j12 * covc[2][2];

    float a  = JC00 * j00 + JC02 * j02 + 0.3f;
    float b  = JC01 * j11 + JC02 * j12;
    float c  = JC11 * j11 + JC12 * j12 + 0.3f;
    float det = a * c - b * b;

    float u = fx * px * z1 + cx;
    float v = fy * py * z1 + cy;

    float inv_det = 1.f / fmaxf(det, EPSF);
    float Ac = c * inv_det, Bc = -b * inv_det, Cc = a * inv_det;

    bool valid = (pz > 0.01f) && (det > EPSF);
    float op = valid ? (1.f / (1.f + __expf(-opac[n]))) : 0.f;

    keys[n] = pz;
    float* P = params + n * PSTRIDE;
    P[0] = u;  P[1] = v;  P[2] = Ac; P[3] = Bc; P[4] = Cc;
    P[5] = op; P[6] = betas[n];
    P[7] = colors[n * 3 + 0]; P[8] = colors[n * 3 + 1]; P[9] = colors[n * 3 + 2];
    P[10] = 0.f; P[11] = 0.f;
}

// ---------------------------------------------------------------------------
// Kernel 2: stable depth rank-sort (N=512, one block) + param scatter
// ---------------------------------------------------------------------------
__global__ void gs_sort(const float* __restrict__ keys,
                        const float* __restrict__ params,
                        float* __restrict__ sorted, int N)
{
    __shared__ float sk[NG];
    int i = threadIdx.x;
    sk[i] = keys[i];
    __syncthreads();
    float ki = sk[i];
    int rank = 0;
    for (int j = 0; j < N; ++j) {
        float kj = sk[j];
        rank += (kj < ki) || (kj == ki && j < i);
    }
    const float* src = params + i * PSTRIDE;
    float*       dst = sorted + rank * PSTRIDE;
    #pragma unroll
    for (int f = 0; f < PSTRIDE; ++f) dst[f] = src[f];
}

// ---------------------------------------------------------------------------
// Kernel 3: tiled render. Block = 256 threads = 8 waves; wave owns a
// 16-pixel strip; WMMA F32 16x16x4 does the w x color blend (rgb + acc).
// The sorted gaussian table (24 KB) is staged into LDS with CDNA5
// async global->LDS copies (ASYNCcnt path), not VGPR staging.
// ---------------------------------------------------------------------------
__global__ __launch_bounds__(256) void gs_render(const float* __restrict__ sp,
                                                 float* __restrict__ out, int N)
{
    __shared__ float lp[NG * PSTRIDE];

    __builtin_prefetch(sp, 0, 3);   // global_prefetch_b8 (warm L2 for the DMA)

    int tid = threadIdx.x;
    const int n4 = NG * PSTRIDE / 4;     // 1536 16-byte packets

    // LDS byte address of lp (addrspace(3) offset) for the async-copy VDST operand
    unsigned lds_base =
        (unsigned)(unsigned long long)(__attribute__((address_space(3))) float*)lp;
    const char* gbase = (const char*)sp;

    for (int idx = tid; idx < n4; idx += 256) {
        unsigned            loff = lds_base + (unsigned)idx * 16u;
        unsigned long long  ga   = (unsigned long long)(gbase + (size_t)idx * 16);
        asm volatile("global_load_async_to_lds_b128 %0, %1, off"
                     :: "v"(loff), "v"(ga) : "memory");
    }
    asm volatile("s_wait_asynccnt 0" ::: "memory");
    __syncthreads();

    int wave = tid >> 5;
    int lane = tid & 31;
    int m    = lane & 15;            // pixel slot within strip (= A-matrix row M, = D-column N)
    bool hi  = lane >= 16;

    int x = blockIdx.x * 16 + m;
    int y = blockIdx.y * 8 + wave;
    float xs = (float)x + 0.5f;
    float ys = (float)y + 0.5f;

    float T = 1.f;                   // running transmittance (identical in lane pair)
    v8f acc = {};                    // D: rows = pixels, cols 0..2 = rgb, col 3 = acc

#if __has_builtin(__builtin_amdgcn_wmma_f32_16x16x4_f32)
    for (int g0 = 0; g0 < N; g0 += 4) {
        float w4[4];
        #pragma unroll
        for (int j = 0; j < 4; ++j) {
            const float* P = &lp[(g0 + j) * PSTRIDE];
            float dx = xs - P[0];
            float dy = ys - P[1];
            float sigma = 0.5f * (P[2] * dx * dx + P[4] * dy * dy) + P[3] * dx * dy;
            sigma = fmaxf(sigma, 1e-8f);
            float alpha = P[5] * __expf(-__powf(sigma, P[6]));
            alpha = fminf(alpha, 0.999f);
            alpha = (alpha > (1.f / 255.f)) ? alpha : 0.f;
            w4[j] = alpha * T;
            T *= (1.f - alpha);
        }
        // A (16x4 f32): lanes 0-15 hold K=0,1; lanes 16-31 hold K=2,3
        v2f A;
        A.x = hi ? w4[2] : w4[0];
        A.y = hi ? w4[3] : w4[1];
        // B (4x16 f32): lane column n = m; rows K=0,1 (lo lanes) / K=2,3 (hi lanes)
        int k0 = g0 + (hi ? 2 : 0);
        int k1 = k0 + 1;
        v2f Bv;
        Bv.x = (m < 3) ? lp[k0 * PSTRIDE + 7 + m] : (m == 3 ? 1.f : 0.f);
        Bv.y = (m < 3) ? lp[k1 * PSTRIDE + 7 + m] : (m == 3 ? 1.f : 0.f);

        acc = __builtin_amdgcn_wmma_f32_16x16x4_f32(false, A, false, Bv,
                                                    (short)0, acc, false, false);
    }
#else
    // Fallback: f16 WMMA, K = 32 gaussians per chunk.
    for (int g0 = 0; g0 < N; g0 += 32) {
        union { v16h v; _Float16 h[16]; } Au;
        union { v16h v; _Float16 h[16]; } Bu;
        #pragma unroll
        for (int k = 0; k < 32; ++k) {
            const float* P = &lp[(g0 + k) * PSTRIDE];
            float dx = xs - P[0];
            float dy = ys - P[1];
            float sigma = 0.5f * (P[2] * dx * dx + P[4] * dy * dy) + P[3] * dx * dy;
            sigma = fmaxf(sigma, 1e-8f);
            float alpha = P[5] * __expf(-__powf(sigma, P[6]));
            alpha = fminf(alpha, 0.999f);
            alpha = (alpha > (1.f / 255.f)) ? alpha : 0.f;
            float wk = alpha * T;
            T *= (1.f - alpha);
            int slot = -1;
            if (!hi) { if (k < 8) slot = k; else if (k >= 16 && k < 24) slot = k - 8; }
            else     { if (k >= 8 && k < 16) slot = k - 8; else if (k >= 24) slot = k - 16; }
            if (slot >= 0) Au.h[slot] = (_Float16)wk;
        }
        #pragma unroll
        for (int r = 0; r < 16; ++r) {
            int k = g0 + (hi ? 16 : 0) + r;
            float cv = (m < 3) ? lp[k * PSTRIDE + 7 + m] : (m == 3 ? 1.f : 0.f);
            Bu.h[r] = (_Float16)cv;
        }
        acc = __builtin_amdgcn_wmma_f32_16x16x32_f16(false, Au.v, false, Bu.v,
                                                     (short)0, acc, false, false);
    }
#endif

    // Epilogue: D lane (col N = m) holds pixels M = r (+8 for hi lanes) in VGPR r.
    if (m < 4) {
        int base = hi ? 8 : 0;
        #pragma unroll
        for (int r = 0; r < 8; ++r) {
            int pm = base + r;
            int xx = blockIdx.x * 16 + pm;
            float val = acc[r];
            if (m < 3) out[(y * WW + xx) * 3 + m] = val;          // rgb
            else       out[HH * WW * 3 + y * WW + xx] = val;      // acc
        }
    }
}

// ---------------------------------------------------------------------------
extern "C" void kernel_launch(void* const* d_in, const int* in_sizes, int n_in,
                              void* d_out, int out_size, void* d_ws, size_t ws_size,
                              hipStream_t stream)
{
    const float* means    = (const float*)d_in[0];
    const float* quats    = (const float*)d_in[1];
    const float* scales   = (const float*)d_in[2];
    const float* opac     = (const float*)d_in[3];
    const float* colors   = (const float*)d_in[4];
    const float* betas    = (const float*)d_in[5];
    const float* viewmats = (const float*)d_in[6];
    const float* Ks       = (const float*)d_in[7];
    int N = in_sizes[3];             // opacities is (N,)

    float* keys   = (float*)d_ws;
    float* params = keys + NG;
    float* sorted = params + NG * PSTRIDE;

    gs_preprocess<<<(N + 255) / 256, 256, 0, stream>>>(means, quats, scales, opac,
                                                       colors, betas, viewmats, Ks,
                                                       keys, params, N);
    gs_sort<<<1, N, 0, stream>>>(keys, params, sorted, N);

    dim3 grid(WW / 16, HH / 8);
    gs_render<<<grid, 256, 0, stream>>>(sorted, (float*)d_out, N);
}
